// EnhancedAttentionLayer_16415365005739
// MI455X (gfx1250) — matrix operand order repaired and checked by reading, not compiled, not run on hardware
//
#include <hip/hip_runtime.h>

#define N_ROWS   100000
#define D_IN     512
#define H_DIM    2048
#define N_HEADS  4
#define SEGS     10000

// ---- per-WG tiling for the fused MLP ----
#define ROWS     80          // rows per workgroup: 100000 = 80 * 1250 exactly
#define HC       256         // hidden-dim chunk processed per iteration (8 chunks)
#define XSP      520         // xs row stride (bf16 elems), mult of 8 -> 16B aligned rows
#define H1CP     264         // h1 chunk row stride (bf16 elems)
#define H2P      516         // h2 row stride (f32 elems)
#define SM_XS_BYTES   (ROWS * XSP * 2)           // 83200
#define SM_H1C_BYTES  (ROWS * H1CP * 2)          // 42240 per buffer
#define SM_BYTES      (SM_XS_BYTES + 2 * SM_H1C_BYTES)   // 167680 (h2f overlays from 0)

typedef __attribute__((ext_vector_type(16))) __bf16 v16bf;
typedef __attribute__((ext_vector_type(8)))  float  v8f;

__device__ __forceinline__ v16bf frag_from(uint4 lo, uint4 hi) {
  union { uint4 u[2]; v16bf v; } t;
  t.u[0] = lo; t.u[1] = hi;
  return t.v;
}

// --------------------------------------------------------------------------
// Pack a row-major f32 weight matrix W[K][N] into bf16 WMMA B-fragment order.
// Tile (nt, kt) of shape K=32 x N=16 is stored as 512 contiguous bf16:
//   lane (0..31) * 16 + j, h = lane>>4: K = kt*32 + (j>>3)*16 + h*8 + (j&7)
//                          (N = lane & 15)
// This matches the ISA 16-bit B-matrix 32x16 VGPR layout exactly, so the GEMM
// inner loop loads each fragment as two 16-byte loads per lane.
// --------------------------------------------------------------------------
__global__ void pack_weights(const float* __restrict__ W,
                             unsigned short* __restrict__ P, int K, int N) {
  int e = blockIdx.x * blockDim.x + threadIdx.x;
  if (e >= K * N) return;
  int NKT    = K >> 5;
  int tile   = e >> 9;
  int within = e & 511;
  int lane   = within >> 4;
  int j      = within & 15;
  int hi     = lane >> 4;
  int nl     = lane & 15;
  int kt     = tile % NKT;
  int nt     = tile / NKT;
  int k      = (kt << 5) + ((j >> 3) << 4) + hi * 8 + (j & 7);
  int n      = (nt << 4) + nl;
  __bf16 bv = (__bf16)W[(size_t)k * N + n];
  P[e] = __builtin_bit_cast(unsigned short, bv);
}

// bias1' = b1 + alpha * W1[512,:]  (folds the concatenated alpha column);
// zero-init segment max (uint bits of positive floats) and segment sums.
__global__ void prep_misc(const float* __restrict__ W1, const float* __restrict__ b1,
                          const float* __restrict__ alpha,
                          float* __restrict__ bias1p,
                          unsigned* __restrict__ segmax, float* __restrict__ segsum) {
  int t = blockIdx.x * blockDim.x + threadIdx.x;
  if (t < H_DIM) bias1p[t] = b1[t] + alpha[0] * W1[(size_t)512 * H_DIM + t];
  if (t < SEGS * N_HEADS) { segmax[t] = 0u; segsum[t] = 0.0f; }
}

// --------------------------------------------------------------------------
// Fused: x -> LDS via async global->LDS DMA, GEMM1(relu) chunked into LDS,
// GEMM2(relu) accumulated in registers, LayerNorm, 4-head projection,
// sigmoid, segment atomicMax.
// 256 threads = 8 waves; wave w owns 64 output cols (GEMM2) / 32 cols per
// chunk (GEMM1); 5 M-tiles cover the 80 rows.
// --------------------------------------------------------------------------
__global__ __launch_bounds__(256) void fused_mlp(
    const float* __restrict__ x, const long long* __restrict__ rowid,
    const unsigned short* __restrict__ W1p_, const float* __restrict__ bias1p,
    const unsigned short* __restrict__ W2p_, const float* __restrict__ b2,
    const float* __restrict__ ln_g, const float* __restrict__ ln_b,
    const float* __restrict__ Wa, const float* __restrict__ ba,
    float* __restrict__ zbuf, unsigned* __restrict__ segmax) {
  __shared__ __align__(16) unsigned char smem[SM_BYTES];
  __bf16* xs  = (__bf16*)smem;                        // [80][520] bf16
  __bf16* h1c = (__bf16*)(smem + SM_XS_BYTES);        // [2][80][264] bf16
  float*  h2f = (float*)smem;                         // [80][516] f32 (overlay)

  const __bf16* W1p = (const __bf16*)W1p_;
  const __bf16* W2p = (const __bf16*)W2p_;

  const int tid  = threadIdx.x;
  const int w    = tid >> 5;
  const int lane = tid & 31;
  const int lrow = lane & 15;
  const int hi   = lane >> 4;
  const int rb   = blockIdx.x * ROWS;

  // ---- stage 0: x tile (f32) -> LDS staging via GLOBAL_LOAD_ASYNC_TO_LDS,
  //      then LDS->LDS convert into the padded bf16 xs layout.
  //      Two halves of 40 rows (40*512*4 = 80 KB staging) reuse the h1c
  //      region, which is not live until the chunk loop below. ----
  {
    const unsigned lds_stage = (unsigned)(uintptr_t)(smem + SM_XS_BYTES);
    const float*   stage     = (const float*)(smem + SM_XS_BYTES);
    for (int half = 0; half < 2; ++half) {
      const char* gbase = (const char*)(x + (size_t)(rb + half * 40) * D_IN);
      for (int i = 0; i < 20; ++i) {
        int s = i * 256 + tid;                  // 16B segment; 5120 per half
        unsigned long long ga = (unsigned long long)(uintptr_t)(gbase + (size_t)s * 16);
        unsigned la = lds_stage + (unsigned)s * 16u;
        asm volatile("global_load_async_to_lds_b128 %0, %1, off"
                     :: "v"(la), "v"(ga) : "memory");
      }
      asm volatile("s_wait_asynccnt 0" ::: "memory");   // this wave's DMA done
      __syncthreads();                                   // everyone's DMA done
      for (int i = 0; i < 20; ++i) {
        int fi = i * 256 + tid;                 // float4 index; 5120 per half
        int r  = fi >> 7;
        int c4 = (fi & 127) << 2;
        float4 v = *(const float4*)(stage + r * D_IN + c4);
        __bf16* dst = xs + (half * 40 + r) * XSP + c4;
        dst[0] = (__bf16)v.x; dst[1] = (__bf16)v.y;
        dst[2] = (__bf16)v.z; dst[3] = (__bf16)v.w;
      }
      __syncthreads();                          // staging free for next half
    }
  }

  v8f acc2[5][4] = {};    // persistent GEMM2 accumulators: 5 M-tiles x 4 N-tiles

  for (int ch = 0; ch < 8; ++ch) {
    __bf16* hbuf = h1c + (ch & 1) * (ROWS * H1CP);

    // ---- GEMM1: h1_chunk[80, 256] = relu(x @ W1[:, ch*256 : +256] + b1') ----
    #pragma unroll
    for (int nt = 0; nt < 2; ++nt) {
      v8f acc[5] = {};
      const int ntg = ch * 16 + w * 2 + nt;     // global N-tile in [0,128)
      for (int kt = 0; kt < 16; ++kt) {
        const __bf16* bp = W1p + ((ntg * 16 + kt) << 9) + lane * 16;
        v16bf b = frag_from(*(const uint4*)bp, *(const uint4*)(bp + 8));
        #pragma unroll
        for (int mt = 0; mt < 5; ++mt) {
          const __bf16* ap = xs + (mt * 16 + lrow) * XSP + kt * 32 + hi * 8;
          v16bf a = frag_from(*(const uint4*)ap, *(const uint4*)(ap + 16));
          acc[mt] = __builtin_amdgcn_wmma_f32_16x16x32_bf16(
              false, a, false, b, (short)0, acc[mt], false, false);
        }
      }
      const int  colL = w * 32 + nt * 16 + lrow;       // chunk-local column
      const float bs  = bias1p[ch * HC + colL];
      #pragma unroll
      for (int mt = 0; mt < 5; ++mt)
        #pragma unroll
        for (int r = 0; r < 8; ++r) {
          int rr = mt * 16 + hi * 8 + r;
          hbuf[rr * H1CP + colL] = (__bf16)fmaxf(acc[mt][r] + bs, 0.0f);
        }
    }
    __syncthreads();

    // ---- GEMM2 partial: acc2 += h1_chunk @ W2[ch*256 : +256, :] ----
    for (int kt = 0; kt < 8; ++kt) {
      v16bf a[5];
      #pragma unroll
      for (int mt = 0; mt < 5; ++mt) {
        const __bf16* ap = hbuf + (mt * 16 + lrow) * H1CP + kt * 32 + hi * 8;
        a[mt] = frag_from(*(const uint4*)ap, *(const uint4*)(ap + 16));
      }
      const int ktg = ch * 8 + kt;                     // global K-tile in [0,64)
      #pragma unroll
      for (int nt = 0; nt < 4; ++nt) {
        const __bf16* bp = W2p + (((w * 4 + nt) * 64 + ktg) << 9) + lane * 16;
        v16bf b = frag_from(*(const uint4*)bp, *(const uint4*)(bp + 8));
        #pragma unroll
        for (int mt = 0; mt < 5; ++mt)
          acc2[mt][nt] = __builtin_amdgcn_wmma_f32_16x16x32_bf16(
              false, a[mt], false, b, (short)0, acc2[mt][nt], false, false);
      }
    }
  }
  __syncthreads();   // last h1 chunk fully consumed; safe to overlay h2f

  // ---- GEMM2 epilogue: bias + relu -> h2f (f32, LDS) ----
  #pragma unroll
  for (int nt = 0; nt < 4; ++nt) {
    const int  col = w * 64 + nt * 16 + lrow;
    const float bs = b2[col];
    #pragma unroll
    for (int mt = 0; mt < 5; ++mt)
      #pragma unroll
      for (int r = 0; r < 8; ++r) {
        int rr = mt * 16 + hi * 8 + r;
        h2f[rr * H2P + col] = fmaxf(acc2[mt][nt][r] + bs, 0.0f);
      }
  }
  __syncthreads();

  // ---- LayerNorm + heads + sigmoid + segment atomicMax; wave w: rows [w*10, w*10+10) ----
  for (int rr = w * 10; rr < w * 10 + 10; ++rr) {
    const float* hr = h2f + rr * H2P;
    float s = 0.0f, s2 = 0.0f;
    #pragma unroll
    for (int i = 0; i < 16; ++i) {
      float v = hr[lane * 16 + i];
      s += v; s2 += v * v;
    }
    #pragma unroll
    for (int off = 16; off > 0; off >>= 1) {
      s  += __shfl_xor(s,  off, 32);
      s2 += __shfl_xor(s2, off, 32);
    }
    const float mu   = s * (1.0f / 512.0f);
    const float var  = s2 * (1.0f / 512.0f) - mu * mu;
    const float rstd = rsqrtf(var + 1e-5f);

    float a0 = 0.f, a1 = 0.f, a2 = 0.f, a3 = 0.f;
    #pragma unroll
    for (int i = 0; i < 16; ++i) {
      int col = lane * 16 + i;
      float hn = (hr[col] - mu) * rstd * ln_g[col] + ln_b[col];
      const float* wa = Wa + col * 4;
      a0 += hn * wa[0]; a1 += hn * wa[1]; a2 += hn * wa[2]; a3 += hn * wa[3];
    }
    #pragma unroll
    for (int off = 16; off > 0; off >>= 1) {
      a0 += __shfl_xor(a0, off, 32);
      a1 += __shfl_xor(a1, off, 32);
      a2 += __shfl_xor(a2, off, 32);
      a3 += __shfl_xor(a3, off, 32);
    }
    if (lane < 4) {
      float raw = (lane == 0) ? a0 : (lane == 1) ? a1 : (lane == 2) ? a2 : a3;
      raw += ba[lane];
      float z = 1.0f / (1.0f + __expf(-raw));     // sigmoid; TEMP = 1
      int g = rb + rr;
      zbuf[g * 4 + lane] = z;
      int rid = (int)rowid[g];
      atomicMax(segmax + rid * 4 + lane, __float_as_uint(z));  // z>0: bit-monotonic
    }
  }
}

__global__ void seg_expsum(const float* __restrict__ zbuf,
                           const long long* __restrict__ rowid,
                           const unsigned* __restrict__ segmax,
                           float* __restrict__ ebuf, float* __restrict__ segsum) {
  int t = blockIdx.x * blockDim.x + threadIdx.x;
  if (t >= N_ROWS * N_HEADS) return;
  int n = t >> 2, h = t & 3;
  int rid = (int)rowid[n];
  float e = __expf(zbuf[t] - __uint_as_float(segmax[rid * 4 + h]));
  ebuf[t] = e;
  atomicAdd(segsum + rid * 4 + h, e);
}

__global__ void finalize(const float* __restrict__ ebuf,
                         const long long* __restrict__ rowid,
                         const float* __restrict__ segsum, float* __restrict__ out) {
  int n = blockIdx.x * blockDim.x + threadIdx.x;
  if (n >= N_ROWS) return;
  int rid = (int)rowid[n];
  float s = 0.0f;
  #pragma unroll
  for (int h = 0; h < 4; ++h) s += ebuf[n * 4 + h] / segsum[rid * 4 + h];
  out[n] = 0.25f * s;
}

extern "C" void kernel_launch(void* const* d_in, const int* in_sizes, int n_in,
                              void* d_out, int out_size, void* d_ws, size_t ws_size,
                              hipStream_t stream) {
  const float*     x     = (const float*)d_in[0];
  const long long* rowid = (const long long*)d_in[1];   // int64 per reference
  const float*     alpha = (const float*)d_in[2];
  const float*     W1    = (const float*)d_in[3];       // [513, 2048]
  const float*     b1    = (const float*)d_in[4];
  const float*     W2    = (const float*)d_in[5];       // [2048, 512]
  const float*     b2    = (const float*)d_in[6];
  const float*     ln_g  = (const float*)d_in[7];
  const float*     ln_b  = (const float*)d_in[8];
  const float*     Wa    = (const float*)d_in[9];       // [512, 4]
  const float*     ba    = (const float*)d_in[10];
  float* out = (float*)d_out;

  char* ws = (char*)d_ws;
  unsigned short* W1p    = (unsigned short*)(ws);                 // 2 MB
  unsigned short* W2p    = (unsigned short*)(ws + 2097152);       // 2 MB
  float*          bias1p = (float*)(ws + 4194304);                // 8 KB
  unsigned*       segmax = (unsigned*)(ws + 4202496);             // 160 KB
  float*          segsum = (float*)(ws + 4362496);                // 160 KB
  float*          zbuf   = (float*)(ws + 4522496);                // 1.6 MB
  float*          ebuf   = (float*)(ws + 6122496);                // 1.6 MB

  pack_weights<<<(512 * 2048 + 255) / 256, 256, 0, stream>>>(W1, W1p, 512, 2048);
  pack_weights<<<(2048 * 512 + 255) / 256, 256, 0, stream>>>(W2, W2p, 2048, 512);
  prep_misc<<<(SEGS * N_HEADS + 255) / 256, 256, 0, stream>>>(W1, b1, alpha,
                                                              bias1p, segmax, segsum);
  fused_mlp<<<N_ROWS / ROWS, 256, 0, stream>>>(x, rowid, W1p, bias1p, W2p, b2,
                                               ln_g, ln_b, Wa, ba, zbuf, segmax);
  seg_expsum<<<(N_ROWS * N_HEADS + 255) / 256, 256, 0, stream>>>(zbuf, rowid, segmax,
                                                                 ebuf, segsum);
  finalize<<<(N_ROWS + 255) / 256, 256, 0, stream>>>(ebuf, rowid, segsum, out);
}
